// GATLayer_40836549050965
// MI455X (gfx1250) — matrix operand order repaired and checked by reading, not compiled, not run on hardware
//
#include <hip/hip_runtime.h>

// ---------------------------------------------------------------------------
// GAT layer, flash-attention style, CDNA5 (gfx1250, wave32, WMMA)
// B=4, N=4096, F_in=256, F_out=128
// ---------------------------------------------------------------------------

#define B_SZ   4
#define N_SZ   4096
#define F_IN   256
#define F_OUT  128
#define ALPHA  0.2f
#define NEG_BIG -9.0e15f

typedef __attribute__((ext_vector_type(2)))  float  v2f;
typedef __attribute__((ext_vector_type(8)))  float  v8f;
typedef __attribute__((ext_vector_type(16))) __bf16 v16bf;

union Bf16x16 {
    v16bf    v;
    unsigned u[8];
};

static __device__ __forceinline__ unsigned short f32_to_bf16(float f) {
    unsigned u = __builtin_bit_cast(unsigned, f);
    unsigned rnd = 0x7FFFu + ((u >> 16) & 1u);   // round-to-nearest-even
    return (unsigned short)((u + rnd) >> 16);
}
static __device__ __forceinline__ unsigned pack_bf16(float lo, float hi) {
    return (unsigned)f32_to_bf16(lo) | ((unsigned)f32_to_bf16(hi) << 16);
}

// ---------------------------------------------------------------------------
// Kernel 0: pack adj (int32 0/1) into bitmask, 268MB -> 8.4MB (L2-resident)
// ---------------------------------------------------------------------------
__global__ void pack_adj_kernel(const int* __restrict__ adj,
                                unsigned* __restrict__ mask,
                                long long nwords) {
    const int lane = threadIdx.x & 31;
    long long wid   = ((long long)blockIdx.x * blockDim.x + threadIdx.x) >> 5;
    long long wstep = ((long long)gridDim.x * blockDim.x) >> 5;
    for (long long w = wid; w < nwords; w += wstep) {
        int v = adj[(w << 5) + lane];                 // coalesced 128B per wave
        unsigned m = __builtin_amdgcn_ballot_w32(v > 0);
        if (lane == 0) mask[w] = m;
    }
}

// ---------------------------------------------------------------------------
// Kernel 1: Wh = h @ W via V_WMMA_F32_16X16X4_F32 (f32-precise, small GEMM)
//           + f1 = Wh.a1, f2 = Wh.a2 (shuffle-reduce + global f32 atomics)
//           + WhT bf16 transposed [F_OUT][B*N] for flash-kernel B operand
// Block = 128 thr (4 waves); wave -> one 16x16 tile. Grid (1024, 2).
// ---------------------------------------------------------------------------
__global__ void wh_kernel(const float* __restrict__ h,
                          const float* __restrict__ W,
                          const float* __restrict__ a,
                          unsigned short* __restrict__ WhT,
                          float* __restrict__ f1,
                          float* __restrict__ f2) {
    const int lane = threadIdx.x & 31;
    const int wave = threadIdx.x >> 5;
    const int m = lane & 15;          // col within tile / row for A
    const int g = lane >> 4;          // half-wave group
    const int rowb = blockIdx.x * 16; // global row base (flattened B*N)
    const int cb   = (blockIdx.y * 4 + wave) * 16;

    v8f acc = {0.f, 0.f, 0.f, 0.f, 0.f, 0.f, 0.f, 0.f};
    const float* hrow = h + (size_t)(rowb + m) * F_IN;

    for (int k0 = 0; k0 < F_IN; k0 += 4) {
        const int k = k0 + 2 * g;
        // A (16x4 f32): lane row m, VGPR v -> K = 2g + v
        v2f A;
        A.x = hrow[k];
        A.y = hrow[k + 1];
        // B (4x16 f32): lane col m, VGPR v -> K = 2g + v
        v2f Bv;
        Bv.x = W[(size_t)k * F_OUT + cb + m];
        Bv.y = W[(size_t)(k + 1) * F_OUT + cb + m];
        acc = __builtin_amdgcn_wmma_f32_16x16x4_f32(
                  false, A, false, Bv, (short)0, acc, false, false);
    }

    // f1/f2 partial dot products (tile covers cols cb..cb+15)
    const float a1 = a[cb + m];
    const float a2 = a[F_OUT + cb + m];
    #pragma unroll
    for (int v = 0; v < 8; ++v) {
        float s1 = acc[v] * a1;
        float s2 = acc[v] * a2;
        #pragma unroll
        for (int off = 8; off >= 1; off >>= 1) {
            s1 += __shfl_xor(s1, off, 32);
            s2 += __shfl_xor(s2, off, 32);
        }
        if (m == 0) {
            const int rg = rowb + v + 8 * g;   // C-layout: VGPR v -> row v+8g
            atomicAdd(&f1[rg], s1);
            atomicAdd(&f2[rg], s2);
        }
    }

    // Store WhT bf16 transposed: WhT[col][row]; lane's 8 rows are contiguous
    unsigned packed[4];
    #pragma unroll
    for (int i = 0; i < 4; ++i)
        packed[i] = pack_bf16(acc[2 * i], acc[2 * i + 1]);
    uint4* dst = (uint4*)&WhT[(size_t)(cb + m) * (B_SZ * N_SZ) + rowb + 8 * g];
    *dst = make_uint4(packed[0], packed[1], packed[2], packed[3]);
}

// ---------------------------------------------------------------------------
// Kernel 2: flash-GAT. Block = (b, 16-row tile), 4 waves split the 4096 cols
// into 32-col chunks. Online softmax + P(bf16) @ WhT via
// V_WMMA_F32_16X16X32_BF16, FA-style cross-wave combine in LDS, elu epilogue.
// ---------------------------------------------------------------------------
#define FK_WAVES 4

__global__ void flash_kernel(const unsigned* __restrict__ maskbuf,
                             const unsigned short* __restrict__ WhT,
                             const float* __restrict__ f1,
                             const float* __restrict__ f2,
                             float* __restrict__ out) {
    __shared__ float sm[FK_WAVES][16];
    __shared__ float sl[FK_WAVES][16];
    __shared__ float accbuf[16][F_OUT];

    const int lane = threadIdx.x & 31;
    const int wave = threadIdx.x >> 5;
    const int m = lane & 15;
    const int g = lane >> 4;
    const int b = blockIdx.y;
    const int rowbase = blockIdx.x * 16;

    for (int i = threadIdx.x; i < 16 * F_OUT; i += blockDim.x)
        (&accbuf[0][0])[i] = 0.f;

    const float  f1r  = f1[b * N_SZ + rowbase + m];
    const float* f2b  = f2 + b * N_SZ;
    const unsigned* mrow = maskbuf + (size_t)(b * N_SZ + rowbase + m) * (N_SZ / 32);
    const unsigned short* whtb = WhT + (size_t)b * N_SZ;

    float mrun = -3.0e38f;
    float lrun = 0.f;
    v8f acc[8];
    #pragma unroll
    for (int t = 0; t < 8; ++t)
        acc[t] = (v8f){0.f, 0.f, 0.f, 0.f, 0.f, 0.f, 0.f, 0.f};

    for (int jc = wave; jc < N_SZ / 32; jc += FK_WAVES) {
        const int colbase = jc * 32;
        const unsigned mw = mrow[jc];           // 32 adjacency bits for row m

        // scores for this lane's 16 K-slots (A-matrix bf16 16x32 layout):
        // k = g*16 + i (contiguous, matches B)
        float p[16];
        float cmax = -3.0e38f;
        #pragma unroll
        for (int i = 0; i < 16; ++i) {
            const int k = g * 16 + i;
            float s = f1r + f2b[colbase + k];
            s = (s > 0.f) ? s : (ALPHA * s);
            if (!((mw >> k) & 1u)) s = NEG_BIG;
            p[i] = s;
            cmax = fmaxf(cmax, s);
        }
        cmax = fmaxf(cmax, __shfl_xor(cmax, 16, 32));   // full row-chunk max
        const float mnew = fmaxf(mrun, cmax);

        float csum = 0.f;
        #pragma unroll
        for (int i = 0; i < 16; ++i) {
            p[i] = __expf(p[i] - mnew);
            csum += p[i];
        }
        csum += __shfl_xor(csum, 16, 32);

        const float scale = __expf(mrun - mnew);
        lrun = lrun * scale + csum;
        mrun = mnew;

        // rescale accumulators: C-layout row = v + 8g; stats live in lane==row
        float scv[8];
        #pragma unroll
        for (int v = 0; v < 8; ++v)
            scv[v] = __shfl(scale, v + 8 * g, 32);
        #pragma unroll
        for (int t = 0; t < 8; ++t)
            #pragma unroll
            for (int v = 0; v < 8; ++v)
                acc[t][v] *= scv[v];

        // pack P to bf16 A operand
        Bf16x16 A;
        #pragma unroll
        for (int v = 0; v < 8; ++v)
            A.u[v] = pack_bf16(p[2 * v], p[2 * v + 1]);

        // 8 output tiles of 16 cols; B (32x16 bf16): lane col m, K = g*16 + 2v..
        // WhT is [F_OUT][B*N] so the 16 K values are contiguous (32B/lane).
        #pragma unroll
        for (int t = 0; t < 8; ++t) {
            const unsigned short* src =
                whtb + (size_t)(t * 16 + m) * (B_SZ * N_SZ) + colbase + g * 16;
            v16bf Bv = *(const v16bf*)src;     // 32B aligned
            acc[t] = __builtin_amdgcn_wmma_f32_16x16x32_bf16(
                         false, A.v, false, Bv, (short)0, acc[t], false, false);
        }
    }

    // --- cross-wave flash combine ---
    if (lane < 16) { sm[wave][lane] = mrun; sl[wave][lane] = lrun; }
    __syncthreads();

    #pragma unroll
    for (int v = 0; v < 8; ++v) {
        const int r = v + 8 * g;
        float M = sm[0][r];
        for (int w = 1; w < FK_WAVES; ++w) M = fmaxf(M, sm[w][r]);
        const float alpha = __expf(sm[wave][r] - M);
        #pragma unroll
        for (int t = 0; t < 8; ++t)
            atomicAdd(&accbuf[r][t * 16 + m], acc[t][v] * alpha);  // ds_add_f32
    }
    __syncthreads();

    // epilogue: divide by row sum, elu, write out (coalesced)
    for (int i = threadIdx.x; i < 16 * F_OUT; i += blockDim.x) {
        const int r = i >> 7;
        const int c = i & (F_OUT - 1);
        float M = sm[0][r];
        for (int w = 1; w < FK_WAVES; ++w) M = fmaxf(M, sm[w][r]);
        float L = 0.f;
        for (int w = 0; w < FK_WAVES; ++w) L += sl[w][r] * __expf(sm[w][r] - M);
        float x = accbuf[r][c] / L;
        x = (x > 0.f) ? x : (__expf(x) - 1.f);
        out[((size_t)b * N_SZ + rowbase + r) * F_OUT + c] = x;
    }
}

// ---------------------------------------------------------------------------
// Launch
// ---------------------------------------------------------------------------
extern "C" void kernel_launch(void* const* d_in, const int* in_sizes, int n_in,
                              void* d_out, int out_size, void* d_ws, size_t ws_size,
                              hipStream_t stream) {
    const float* h   = (const float*)d_in[0];
    const int*   adj = (const int*)d_in[1];
    const float* W   = (const float*)d_in[2];
    const float* a   = (const float*)d_in[3];
    float*       out = (float*)d_out;

    // workspace: maskbuf 8MB | WhT bf16 4MB | f1 64KB | f2 64KB  (~12.2MB)
    char* ws = (char*)d_ws;
    unsigned*       maskbuf = (unsigned*)ws;
    unsigned short* WhT     = (unsigned short*)(ws + (size_t)8 * 1024 * 1024);
    float*          f1      = (float*)(ws + (size_t)12 * 1024 * 1024);
    float*          f2      = f1 + B_SZ * N_SZ;

    hipMemsetAsync(f1, 0, (size_t)2 * B_SZ * N_SZ * sizeof(float), stream);

    const long long nwords = (long long)B_SZ * N_SZ * (N_SZ / 32);
    pack_adj_kernel<<<2048, 256, 0, stream>>>(adj, maskbuf, nwords);

    wh_kernel<<<dim3((B_SZ * N_SZ) / 16, F_OUT / 64), 128, 0, stream>>>(
        h, W, a, WhT, f1, f2);

    flash_kernel<<<dim3(N_SZ / 16, B_SZ), 32 * FK_WAVES, 0, stream>>>(
        maskbuf, WhT, f1, f2, out);
}
